// EngramLayer_26774826123654
// MI455X (gfx1250) — matrix operand order repaired and checked by reading, not compiled, not run on hardware
//
#include <hip/hip_runtime.h>
#include <hip/hip_bf16.h>
#include <math.h>

#define B_SZ 4
#define T_SZ 4096
#define D_SZ 2048
#define DMEM 1024
#define H_SZ 16
#define DHEAD 64
#define TABLE_SZ 131072
#define KCONV 4
#define DILATION 2
#define NTOK (B_SZ * T_SZ)         // 16384
#define EPSQK 1.1920929e-7f        // np.finfo(float32).eps
#define EPSCONV 1e-5f

typedef __bf16 bf16;
typedef __attribute__((ext_vector_type(16))) __bf16 bf16x16;
typedef __attribute__((ext_vector_type(8)))  __bf16 bf16x8;
typedef __attribute__((ext_vector_type(4)))  __bf16 bf16x4;
typedef __attribute__((ext_vector_type(8)))  float  f32x8;
typedef __attribute__((ext_vector_type(4)))  unsigned int u32x4;
typedef __attribute__((ext_vector_type(8)))  unsigned int u32x8;

// ---------------------------------------------------------------------------
// Kernel 1: convert key_W / value_W (f32 [D][DMEM]) -> bf16, same layout.
// ---------------------------------------------------------------------------
__global__ __launch_bounds__(256)
void convert_w_kernel(const float* __restrict__ kW, const float* __restrict__ vW,
                      bf16* __restrict__ kWb, bf16* __restrict__ vWb) {
    size_t i = ((size_t)blockIdx.x * blockDim.x + threadIdx.x) * 4;
    float4 a = *(const float4*)(kW + i);
    float4 b = *(const float4*)(vW + i);
    bf16x4 oa, ob;
    oa[0] = (bf16)a.x; oa[1] = (bf16)a.y; oa[2] = (bf16)a.z; oa[3] = (bf16)a.w;
    ob[0] = (bf16)b.x; ob[1] = (bf16)b.y; ob[2] = (bf16)b.z; ob[3] = (bf16)b.w;
    *(bf16x4*)(kWb + i) = oa;
    *(bf16x4*)(vWb + i) = ob;
}

// ---------------------------------------------------------------------------
// Kernel 2: gather hash embeddings -> e_bf16 [NTOK][DMEM]
// ---------------------------------------------------------------------------
__global__ __launch_bounds__(256)
void gather_kernel(const int* __restrict__ hash_ids, const float* __restrict__ emb,
                   bf16* __restrict__ e_out) {
    size_t i   = ((size_t)blockIdx.x * blockDim.x + threadIdx.x) * 4;
    int  token = (int)(i / DMEM);
    int  m     = (int)(i % DMEM);
    int  h     = m >> 6;
    int  id    = hash_ids[token * H_SZ + h];
    size_t src = ((size_t)(h * TABLE_SZ + id)) * DHEAD + (m & 63);
    float4 a = *(const float4*)(emb + src);
    bf16x4 o;
    o[0] = (bf16)a.x; o[1] = (bf16)a.y; o[2] = (bf16)a.z; o[3] = (bf16)a.w;
    *(bf16x4*)(e_out + i) = o;
}

// ---------------------------------------------------------------------------
// Kernel 3: fused K/V projection GEMM with v_wmma_f32_16x16x32_bf16.
// Double-buffered LDS; A via GLOBAL_LOAD_ASYNC_TO_LDS_B128 (ASYNCcnt),
// B tiles via TDM TENSOR_LOAD_TO_LDS (TENSORcnt, wave 0 only).
// The smem base pointer is passed into every asm so LLVM sees the LDS
// object escape and keeps the ds_load fragment reads live.
// ---------------------------------------------------------------------------
#define MT 64
#define NT 128
#define KT 32
#define SBK_OFF 4096u                    // A: 64*32*2 = 4096 B at offset 0
#define SBV_OFF 12288u                   // Bk: 128*32*2 = 8192 B
#define BUF_BYTES 20480u                 // per double-buffer stage
#define SMEM_BYTES (2u * BUF_BYTES)      // 40 KB

__device__ __forceinline__ void async_b128_to_lds(unsigned lds_byte_off, const bf16* gaddr,
                                                  const bf16* smembase) {
    asm volatile("global_load_async_to_lds_b128 %0, %1, off"
                 :: "v"(lds_byte_off), "v"(gaddr), "v"(smembase)
                 : "memory");
}

__device__ __forceinline__ void tdm_load_tile(unsigned lds_byte_off, const bf16* gaddr,
                                              u32x8 g1, u32x4 gz, const bf16* smembase) {
    unsigned long long ga = (unsigned long long)(uintptr_t)gaddr;
    u32x4 g0;
    g0[0] = 1u;                                        // count=1, user mode
    g0[1] = lds_byte_off;                              // lds_addr
    g0[2] = (unsigned)(ga & 0xFFFFFFFFull);            // global_addr[31:0]
    g0[3] = (unsigned)((ga >> 32) & 0x1FFFFFFull)      // global_addr[56:32]
          | 0x80000000u;                               // type=2 ("image")
    asm volatile("tensor_load_to_lds %0, %1, %2, %3"
                 :: "s"(g0), "s"(g1), "s"(gz), "s"(gz), "v"(smembase)
                 : "memory");
}

__global__ __launch_bounds__(256)
void gemm_kv_kernel(const bf16* __restrict__ E, const bf16* __restrict__ Wk,
                    const bf16* __restrict__ Wv, const float* __restrict__ keyb,
                    const float* __restrict__ valb,
                    float* __restrict__ Kout, float* __restrict__ Vout) {
    __shared__ __align__(16) bf16 smem[SMEM_BYTES / 2];

    const int tid  = threadIdx.x;
    const int lane = tid & 31;
    const int wave = tid >> 5;
    const int wm   = wave & 1;        // 0..1 -> 32-row slice
    const int wn   = wave >> 1;       // 0..3 -> 32-col slice
    const int m0   = blockIdx.y * MT;
    const int n0   = blockIdx.x * NT;

    const int lr = tid >> 2;          // 0..63  (A staging row)
    const int lc = (tid & 3) * 8;     // 0,8,16,24 (A staging K chunk)

    const int lrow = lane & 15;
    const int ka8  = (lane >> 4) * 8;    // unused; A frags also use kb16
    const int kb16 = (lane >> 4) * 16;   // B-frag K base

    f32x8 zero = {0.f, 0.f, 0.f, 0.f, 0.f, 0.f, 0.f, 0.f};
    f32x8 acck[2][2], accv[2][2];
#pragma unroll
    for (int mi = 0; mi < 2; ++mi)
#pragma unroll
        for (int ni = 0; ni < 2; ++ni) { acck[mi][ni] = zero; accv[mi][ni] = zero; }

    // loop-invariant TDM descriptor words (group1, groups 2/3)
    u32x8 g1c;
    g1c[0] = 0x00010000u;                 // data_size=1 (2 bytes), workgroup_mask=0
    g1c[1] = (unsigned)DMEM << 16;        // tensor_dim0 = 1024
    g1c[2] = (unsigned)D_SZ << 16;        // tensor_dim1[15:0] = 2048
    g1c[3] = (unsigned)KT << 16;          // tile_dim0 = 32 (K)
    g1c[4] = (unsigned)NT;                // tile_dim1 = 128 rows
    g1c[5] = (unsigned)DMEM;              // tensor_dim0_stride = 1024
    g1c[6] = 0u;
    g1c[7] = 0u;
    u32x4 gzc; gzc[0] = 0u; gzc[1] = 0u; gzc[2] = 0u; gzc[3] = 0u;

    const unsigned ldsA = (unsigned)(lr * (KT * 2) + lc * 2);   // within a buffer

    // ---- prologue: fill buffer 0 ----
    async_b128_to_lds(ldsA, E + (size_t)(m0 + lr) * DMEM + lc, smem);
    if (wave == 0) {
        tdm_load_tile(SBK_OFF, Wk + (size_t)n0 * DMEM, g1c, gzc, smem);
        tdm_load_tile(SBV_OFF, Wv + (size_t)n0 * DMEM, g1c, gzc, smem);
        __builtin_amdgcn_s_wait_tensorcnt(0);
    }
    asm volatile("s_wait_asynccnt 0" ::: "memory");
    __syncthreads();

    unsigned cur = 0;
    for (int kk = 0; kk < DMEM; kk += KT) {
        // ---- issue DMA for NEXT tile into the other buffer (overlaps WMMA) ----
        int kn = kk + KT;
        if (kn < DMEM) {
            unsigned bo = (cur ^ 1u) * BUF_BYTES;
            async_b128_to_lds(bo + ldsA, E + (size_t)(m0 + lr) * DMEM + kn + lc, smem);
            if (wave == 0) {
                tdm_load_tile(bo + SBK_OFF, Wk + (size_t)n0 * DMEM + kn, g1c, gzc, smem);
                tdm_load_tile(bo + SBV_OFF, Wv + (size_t)n0 * DMEM + kn, g1c, gzc, smem);
            }
        }

        const bf16* sA  = smem + (size_t)cur * (BUF_BYTES / 2);
        const bf16* sBk = sA + SBK_OFF / 2;
        const bf16* sBv = sA + (SBV_OFF - SBK_OFF + SBK_OFF) / 2;  // == sA + SBV_OFF/2

        // --- A fragments: lanes 0-15 K[0..16), lanes 16-31 K[16..32)
        bf16x16 afrag[2], bkfrag[2], bvfrag[2];
#pragma unroll
        for (int mi = 0; mi < 2; ++mi) {
            int row = wm * 32 + mi * 16 + lrow;
            bf16x8 lo = *(const bf16x8*)&sA[row * KT + kb16];
            bf16x8 hi = *(const bf16x8*)&sA[row * KT + kb16 + 8];
            afrag[mi] = __builtin_shufflevector(lo, hi,
                0,1,2,3,4,5,6,7,8,9,10,11,12,13,14,15);
        }
        // --- B fragments: column d = lane&15, 16 contiguous K per lane-half
#pragma unroll
        for (int ni = 0; ni < 2; ++ni) {
            int col = wn * 32 + ni * 16 + lrow;
            bf16x8 klo = *(const bf16x8*)&sBk[col * KT + kb16];
            bf16x8 khi = *(const bf16x8*)&sBk[col * KT + kb16 + 8];
            bkfrag[ni] = __builtin_shufflevector(klo, khi,
                0,1,2,3,4,5,6,7,8,9,10,11,12,13,14,15);
            bf16x8 vlo = *(const bf16x8*)&sBv[col * KT + kb16];
            bf16x8 vhi = *(const bf16x8*)&sBv[col * KT + kb16 + 8];
            bvfrag[ni] = __builtin_shufflevector(vlo, vhi,
                0,1,2,3,4,5,6,7,8,9,10,11,12,13,14,15);
        }

        // --- 8 WMMAs per K step per wave (A-fragments shared by K and V) ---
#pragma unroll
        for (int mi = 0; mi < 2; ++mi)
#pragma unroll
            for (int ni = 0; ni < 2; ++ni) {
                acck[mi][ni] = __builtin_amdgcn_wmma_f32_16x16x32_bf16(
                    false, afrag[mi], false, bkfrag[ni], (short)0, acck[mi][ni], false, false);
                accv[mi][ni] = __builtin_amdgcn_wmma_f32_16x16x32_bf16(
                    false, afrag[mi], false, bvfrag[ni], (short)0, accv[mi][ni], false, false);
            }

        // ---- wait for the NEXT buffer's DMA, then swap ----
        asm volatile("s_wait_asynccnt 0" ::: "memory");
        if (wave == 0) __builtin_amdgcn_s_wait_tensorcnt(0);
        __syncthreads();
        cur ^= 1u;
    }

    // --- epilogue: C layout: VGPR r -> M = r + (lane>=16)*8, N = lane&15 ---
    const int mofs = (lane >> 4) * 8;
    const int ncol = lane & 15;
#pragma unroll
    for (int mi = 0; mi < 2; ++mi)
#pragma unroll
        for (int ni = 0; ni < 2; ++ni) {
            int colg = n0 + wn * 32 + ni * 16 + ncol;
            float bk = keyb[colg], bv = valb[colg];
#pragma unroll
            for (int r = 0; r < 8; ++r) {
                int rowg = m0 + wm * 32 + mi * 16 + mofs + r;
                Kout[(size_t)rowg * D_SZ + colg] = acck[mi][ni][r] + bk;
                Vout[(size_t)rowg * D_SZ + colg] = accv[mi][ni][r] + bv;
            }
        }
}

// ---------------------------------------------------------------------------
// Kernel 4: per-token fused RMSNorm(q)/RMSNorm(k) dot gate + gated_v
// + conv-RMSNorm statistic (mean(gated_v^2) = gate^2 * mean(v^2)).
// ---------------------------------------------------------------------------
__global__ __launch_bounds__(256)
void gate_kernel(const float* __restrict__ hs, const float* __restrict__ Kmat,
                 const float* __restrict__ Vmat, const float* __restrict__ qg,
                 const float* __restrict__ kg,
                 float* __restrict__ gated_v, float* __restrict__ rstd_conv) {
    const int tok = blockIdx.x;
    const int tid = threadIdx.x;
    const float* h = hs   + (size_t)tok * D_SZ;
    const float* k = Kmat + (size_t)tok * D_SZ;
    const float* v = Vmat + (size_t)tok * D_SZ;

    __shared__ float red0[256], red1[256], red2[256], red3[256];
    float sh2 = 0.f, sk2 = 0.f, shk = 0.f, sv2 = 0.f;
    for (int d = tid; d < D_SZ; d += 256) {
        float hv = h[d], kv = k[d], vv = v[d];
        sh2 += hv * hv;
        sk2 += kv * kv;
        shk += (hv * qg[d]) * (kv * kg[d]);
        sv2 += vv * vv;
    }
    red0[tid] = sh2; red1[tid] = sk2; red2[tid] = shk; red3[tid] = sv2;
    __syncthreads();
    for (int s = 128; s > 0; s >>= 1) {
        if (tid < s) {
            red0[tid] += red0[tid + s];
            red1[tid] += red1[tid + s];
            red2[tid] += red2[tid + s];
            red3[tid] += red3[tid + s];
        }
        __syncthreads();
    }
    __shared__ float gate_s, rstd_s;
    if (tid == 0) {
        float invD = 1.0f / (float)D_SZ;
        float rh  = rsqrtf(red0[0] * invD + EPSQK);
        float rk  = rsqrtf(red1[0] * invD + EPSQK);
        float dot = red2[0] * rh * rk * (1.0f / sqrtf((float)D_SZ));
        float sgn = (dot > 0.f) ? 1.f : ((dot < 0.f) ? -1.f : 0.f);
        dot = sqrtf(fmaxf(fabsf(dot), 1e-6f)) * sgn;
        float g = 1.f / (1.f + expf(-dot));
        gate_s = g;
        rstd_s = rsqrtf(g * g * (red3[0] * invD) + EPSCONV);
    }
    __syncthreads();
    float g = gate_s;
    for (int d = tid; d < D_SZ; d += 256)
        gated_v[(size_t)tok * D_SZ + d] = g * v[d];
    if (tid == 0) rstd_conv[tok] = rstd_s;
}

// ---------------------------------------------------------------------------
// Kernel 5: depthwise dilated causal conv (K=4, dil=2) + SiLU + residual.
// ---------------------------------------------------------------------------
__global__ __launch_bounds__(256)
void conv_kernel(const float* __restrict__ gated_v, const float* __restrict__ rstd,
                 const float* __restrict__ cg, const float* __restrict__ cw,
                 float* __restrict__ out) {
    size_t idx = (size_t)blockIdx.x * 256 + threadIdx.x;
    int d     = (int)(idx % D_SZ);
    int tglob = (int)(idx / D_SZ);
    int b = tglob / T_SZ, t = tglob % T_SZ;
    float gd  = cg[d];
    float acc = 0.f;
#pragma unroll
    for (int j = 0; j < KCONV; ++j) {
        int tt = t - (KCONV - 1 - j) * DILATION;
        if (tt >= 0) {
            size_t src = ((size_t)(b * T_SZ + tt)) * D_SZ + d;
            acc += cw[d * KCONV + j] * (gated_v[src] * rstd[b * T_SZ + tt] * gd);
        }
    }
    float yc = acc / (1.f + expf(-acc));     // SiLU
    out[idx] = yc + gated_v[idx];
}

// ---------------------------------------------------------------------------
extern "C" void kernel_launch(void* const* d_in, const int* in_sizes, int n_in,
                              void* d_out, int out_size, void* d_ws, size_t ws_size,
                              hipStream_t stream) {
    const float* hidden = (const float*)d_in[0];
    const int*   hash   = (const int*)  d_in[1];
    const float* emb    = (const float*)d_in[2];
    const float* keyW   = (const float*)d_in[3];
    const float* keyb   = (const float*)d_in[4];
    const float* valW   = (const float*)d_in[5];
    const float* valb   = (const float*)d_in[6];
    const float* qg     = (const float*)d_in[7];
    const float* kg     = (const float*)d_in[8];
    const float* cg     = (const float*)d_in[9];
    const float* cw     = (const float*)d_in[10];
    float* out = (float*)d_out;

    char* ws = (char*)d_ws;
    bf16*  Wkb = (bf16*)(ws);                       //   4 MB
    bf16*  Wvb = (bf16*)(ws + (4ull  << 20));       //   4 MB
    bf16*  Eb  = (bf16*)(ws + (8ull  << 20));       //  32 MB
    float* Kf  = (float*)(ws + (40ull << 20));      // 128 MB
    float* Vf  = (float*)(ws + (168ull << 20));     // 128 MB
    float* GV  = (float*)(ws + (296ull << 20));     // 128 MB
    float* RS  = (float*)(ws + (424ull << 20));     //  64 KB

    convert_w_kernel<<<(D_SZ * DMEM / 4) / 256, 256, 0, stream>>>(keyW, valW, Wkb, Wvb);
    gather_kernel<<<(NTOK * DMEM / 4) / 256, 256, 0, stream>>>(hash, emb, Eb);

    dim3 gg(D_SZ / NT, NTOK / MT);   // (16, 256)
    gemm_kv_kernel<<<gg, 256, 0, stream>>>(Eb, Wkb, Wvb, keyb, valb, Kf, Vf);

    gate_kernel<<<NTOK, 256, 0, stream>>>(hidden, Kf, Vf, qg, kg, GV, RS);

    conv_kernel<<<(NTOK * (D_SZ / 256)), 256, 0, stream>>>(GV, RS, cg, cw, out);
}
